// Sparse_AutoEnc_TopK_5050881540814
// MI455X (gfx1250) — matrix operand order repaired and checked by reading, not compiled, not run on hardware
//
#include <hip/hip_runtime.h>
#include <hip/hip_bf16.h>

#define BATCH 8192
#define INF_  768
#define OUTF  24576
#define TOPK  128

typedef __attribute__((ext_vector_type(16))) __bf16 v16bf;
typedef __attribute__((ext_vector_type(8)))  __bf16 v8bf;
typedef __attribute__((ext_vector_type(4)))  __bf16 v4bf;
typedef __attribute__((ext_vector_type(8)))  float  v8f;

// ---------------------------------------------------------------------------
// Kernel 0: f32 -> bf16 pre-conversion (x and We are heavily reused by the
// GEMM: converting once halves staging traffic and strips cvt from hot loop).
// One float4 -> v4bf (8B store) per thread.
// ---------------------------------------------------------------------------
__global__ __launch_bounds__(256) void to_bf16(const float* __restrict__ src,
                                               __bf16* __restrict__ dst)
{
    const size_t i = (size_t)blockIdx.x * 256 + threadIdx.x;   // float4 index
    float4 f = ((const float4*)src)[i];
    v4bf h = { (__bf16)f.x, (__bf16)f.y, (__bf16)f.z, (__bf16)f.w };
    *(v4bf*)(dst + i * 4) = h;
}

// ---------------------------------------------------------------------------
// Kernel 1: encoder GEMM  emb = x @ We^T + be   (bf16 WMMA, f32 accumulate)
// Block = 256 threads = 8 waves. Tile: 128(M) x 128(N) x 32(K).
// Double-buffered LDS, staged with GLOBAL_LOAD_ASYNC_TO_LDS_B128 (ASYNCcnt),
// one barrier per K-step. Wave w computes rows [16w,16w+16) x 128 cols.
// ---------------------------------------------------------------------------
__global__ __launch_bounds__(256) void enc_gemm(const __bf16* __restrict__ xbf,
                                                const __bf16* __restrict__ Webf,
                                                const float* __restrict__ be,
                                                float* __restrict__ emb)
{
    // dynamic LDS: 2 buffers x (A: 128x32 bf16 = 8KB | B: 8KB) = 32KB
    extern __shared__ __bf16 smem[];

    const int tid  = threadIdx.x;
    const int wave = tid >> 5;
    const int lane = tid & 31;
    const int l15  = lane & 15;
    const int lh   = lane >> 4;
    const int bRow = blockIdx.y * 128;   // batch rows
    const int bCol = blockIdx.x * 128;   // output features

    const unsigned ldsBase = __builtin_amdgcn_groupstaticsize();
    const unsigned long long xAddr = (unsigned long long)xbf;
    const unsigned long long wAddr = (unsigned long long)Webf;

    v8f acc[8] = {};

    // Async-stage one 128x32 bf16 tile pair into buffer `buf`.
    // 512 b128 lines per tile; each thread issues 2 A + 2 B async loads.
    auto stage = [&](int kk, int buf) {
        const unsigned aBuf = ldsBase + (unsigned)buf * 16384u;
        const unsigned bBuf = aBuf + 8192u;
#pragma unroll
        for (int j = 0; j < 2; j++) {
            const int idx = j * 256 + tid;
            const int r   = idx >> 2;          // row within tile
            const int sg  = idx & 3;           // 8-elem segment within K=32
            const unsigned ldsOff = (unsigned)((r * 32 + sg * 8) * 2);
            const unsigned aG = (unsigned)(((bRow + r) * INF_ + kk + sg * 8) * 2);
            const unsigned aL = aBuf + ldsOff;
            asm volatile("global_load_async_to_lds_b128 %0, %1, %2"
                         :: "v"(aL), "v"(aG), "s"(xAddr) : "memory");
            const unsigned bG = (unsigned)(((bCol + r) * INF_ + kk + sg * 8) * 2);
            const unsigned bL = bBuf + ldsOff;
            asm volatile("global_load_async_to_lds_b128 %0, %1, %2"
                         :: "v"(bL), "v"(bG), "s"(wAddr) : "memory");
        }
    };

    stage(0, 0);

    for (int i = 0; i < INF_ / 32; i++) {
        // finish our outstanding async stage; barrier covers all waves' stages
        // AND guarantees every wave is done reading the other buffer.
        asm volatile("s_wait_asynccnt 0x0" ::: "memory");
        __syncthreads();
        if (i + 1 < INF_ / 32) stage((i + 1) * 32, (i + 1) & 1);

        const __bf16* As = smem + (i & 1) * 8192;
        const __bf16* Bs = As + 4096;

        // A fragment (16x32 bf16): lane l15 = M row; elems 0-7 = K lh*8..+7,
        // elems 8-15 = K 16+lh*8..+7  (ISA 7.12.2 16-bit A layout)
        const int aOff = (wave * 16 + l15) * 32;
        v8bf a0 = *(const v8bf*)(As + aOff + lh * 8);
        v8bf a1 = *(const v8bf*)(As + aOff + 16 + lh * 8);
        v16bf afrag = __builtin_shufflevector(a0, a1,
            0,1,2,3,4,5,6,7,8,9,10,11,12,13,14,15);

#pragma unroll
        for (int t = 0; t < 8; t++) {
            // B fragment (32x16 bf16): lane l15 = N col; elems 0-7 = K lh*8..+7,
            // elems 8-15 = K 16+lh*8..+7
            const int bOff = (t * 16 + l15) * 32 + lh * 8;
            v8bf b0 = *(const v8bf*)(Bs + bOff);
            v8bf b1 = *(const v8bf*)(Bs + bOff + 16);
            v16bf bfrag = __builtin_shufflevector(b0, b1,
                0,1,2,3,4,5,6,7,8,9,10,11,12,13,14,15);
            acc[t] = __builtin_amdgcn_wmma_f32_16x16x32_bf16(
                false, afrag, false, bfrag, (short)0, acc[t], false, false);
        }
    }

    // Epilogue: C/D layout -> lane l15 = N, VGPR r -> M = lh*8 + r
#pragma unroll
    for (int t = 0; t < 8; t++) {
        const int gc   = bCol + t * 16 + l15;
        const float bv = be[gc];
        const int  rb  = bRow + wave * 16 + lh * 8;
#pragma unroll
        for (int r = 0; r < 8; r++)
            emb[(size_t)(rb + r) * OUTF + gc] = acc[t][r] + bv;
    }
}

// ---------------------------------------------------------------------------
// Kernel 2: per-row exact top-128 via 4-pass radix select on order-transformed
// keys held in LDS (96KB dynamic). Masks non-top-k to zero in place and emits
// compacted (idx,val) pairs for the sparse decoder.
// ---------------------------------------------------------------------------
__device__ __forceinline__ unsigned f2ord(unsigned b) {
    return (b & 0x80000000u) ? ~b : (b | 0x80000000u);
}
__device__ __forceinline__ float ord2f(unsigned u) {
    unsigned b = (u & 0x80000000u) ? (u ^ 0x80000000u) : ~u;
    return __uint_as_float(b);
}

__global__ __launch_bounds__(256) void topk_mask(float* __restrict__ enc,
                                                 int* __restrict__ wsIdx,
                                                 float* __restrict__ wsVal)
{
    extern __shared__ unsigned sKeys[];     // OUTF keys (96 KB)
    __shared__ unsigned hist[256];
    __shared__ unsigned ctrl[4];            // 0:selBin 1:newK 2:eqTaken 3:slot

    const int tid = threadIdx.x;
    const int row = blockIdx.x;
    float* rowp = enc + (size_t)row * OUTF;

    // Load + transform (coalesced float4)
    const float4* rp4 = (const float4*)rowp;
#pragma unroll
    for (int i = 0; i < 24; i++) {
        const int i4 = i * 256 + tid;
        float4 f = rp4[i4];
        uint4 u;
        u.x = f2ord(__float_as_uint(f.x));
        u.y = f2ord(__float_as_uint(f.y));
        u.z = f2ord(__float_as_uint(f.z));
        u.w = f2ord(__float_as_uint(f.w));
        *(uint4*)(sKeys + i4 * 4) = u;
    }
    if (tid < 4) ctrl[tid] = 0u;
    __syncthreads();

    unsigned prefix = 0, k = TOPK;
    for (int pass = 0; pass < 4; pass++) {
        const int shift = 24 - pass * 8;
        hist[tid] = 0u;
        __syncthreads();
        const unsigned pmask = (pass == 0) ? 0u : (0xFFFFFFFFu << (shift + 8));
        for (int i = 0; i < 96; i++) {
            unsigned u = sKeys[i * 256 + tid];
            if ((u & pmask) == prefix)
                atomicAdd(&hist[(u >> shift) & 255u], 1u);
        }
        __syncthreads();
        if (tid == 0) {
            unsigned cum = 0; int sel = 0;
            for (int b = 255; b >= 0; b--) {
                unsigned c = hist[b];
                if (cum + c >= k) { sel = b; break; }
                cum += c;
            }
            ctrl[0] = (unsigned)sel;
            ctrl[1] = k - cum;
        }
        __syncthreads();
        prefix |= ctrl[0] << shift;
        k = ctrl[1];
        __syncthreads();
    }

    const unsigned T = prefix;      // exact 128th-largest key
    const unsigned quota = k;       // how many ties at T to keep
    for (int i = 0; i < 96; i++) {
        const int e = i * 256 + tid;
        const unsigned u = sKeys[e];
        bool keep;
        if (u > T)       keep = true;
        else if (u == T) keep = (atomicAdd(&ctrl[2], 1u) < quota);
        else             keep = false;
        if (keep) {
            unsigned s = atomicAdd(&ctrl[3], 1u);
            wsIdx[(size_t)row * TOPK + s] = e;
            wsVal[(size_t)row * TOPK + s] = ord2f(u);
        } else {
            rowp[e] = 0.0f;   // kept entries already hold correct emb values
        }
    }
}

// ---------------------------------------------------------------------------
// Kernel 3: transpose Wd [768,24576] -> WdT [24576,768] so sparse decode rows
// are contiguous (WdT is L2-resident at 75MB / 192MB L2).
// ---------------------------------------------------------------------------
__global__ __launch_bounds__(256) void transpose_wd(const float* __restrict__ Wd,
                                                    float* __restrict__ WdT)
{
    __shared__ float tile[32][33];
    const int tx = threadIdx.x, ty = threadIdx.y;
    const int gx = blockIdx.x * 32;   // over OUTF
    const int gy = blockIdx.y * 32;   // over INF_
#pragma unroll
    for (int i = 0; i < 32; i += 8)
        tile[ty + i][tx] = Wd[(size_t)(gy + ty + i) * OUTF + gx + tx];
    __syncthreads();
#pragma unroll
    for (int i = 0; i < 32; i += 8)
        WdT[(size_t)(gx + ty + i) * INF_ + gy + tx] = tile[tx][ty + i];
}

// ---------------------------------------------------------------------------
// Kernel 4: sparse decode  dec[b,:] = bd + sum_j val_j * WdT[idx_j,:]
// One block per row; thread t owns outputs {t, t+256, t+512}; all WdT reads
// fully coalesced.
// ---------------------------------------------------------------------------
__global__ __launch_bounds__(256) void dec_sparse(const int* __restrict__ wsIdx,
                                                  const float* __restrict__ wsVal,
                                                  const float* __restrict__ WdT,
                                                  const float* __restrict__ bd,
                                                  float* __restrict__ dec)
{
    __shared__ int   sIdx[TOPK];
    __shared__ float sVal[TOPK];
    const int tid = threadIdx.x;
    const int row = blockIdx.x;
    if (tid < TOPK) {
        sIdx[tid] = wsIdx[(size_t)row * TOPK + tid];
        sVal[tid] = wsVal[(size_t)row * TOPK + tid];
    }
    __syncthreads();
    float a0 = bd[tid], a1 = bd[tid + 256], a2 = bd[tid + 512];
#pragma unroll 4
    for (int j = 0; j < TOPK; j++) {
        const float* wr = WdT + (size_t)sIdx[j] * INF_;
        const float v = sVal[j];
        a0 += v * wr[tid];
        a1 += v * wr[tid + 256];
        a2 += v * wr[tid + 512];
    }
    float* dr = dec + (size_t)row * INF_;
    dr[tid] = a0; dr[tid + 256] = a1; dr[tid + 512] = a2;
}

// ---------------------------------------------------------------------------
extern "C" void kernel_launch(void* const* d_in, const int* in_sizes, int n_in,
                              void* d_out, int out_size, void* d_ws, size_t ws_size,
                              hipStream_t stream)
{
    const float* x  = (const float*)d_in[0];
    const float* We = (const float*)d_in[1];
    const float* be = (const float*)d_in[2];
    const float* Wd = (const float*)d_in[3];
    const float* bd = (const float*)d_in[4];

    // Output layout: decoded [B,768] | encoded [B,24576] | x copy [B,768]
    float* dec = (float*)d_out;
    float* enc = dec + (size_t)BATCH * INF_;
    float* xcp = enc + (size_t)BATCH * OUTF;

    // Workspace: WdT (75MB) | wsIdx (4MB) | wsVal (4MB) | xbf (12.6MB) | Webf (37.7MB)
    char* ws = (char*)d_ws;
    float*  WdT   = (float*)ws;                    ws += (size_t)OUTF * INF_ * sizeof(float);
    int*    wsIdx = (int*)ws;                      ws += (size_t)BATCH * TOPK * sizeof(int);
    float*  wsVal = (float*)ws;                    ws += (size_t)BATCH * TOPK * sizeof(float);
    __bf16* xbf   = (__bf16*)ws;                   ws += (size_t)BATCH * INF_ * sizeof(__bf16);
    __bf16* Webf  = (__bf16*)ws;

    // 0. One-shot f32 -> bf16 conversion of the reused GEMM operands.
    to_bf16<<<(BATCH * INF_) / (4 * 256), 256, 0, stream>>>(x, xbf);
    to_bf16<<<(OUTF * INF_) / (4 * 256), 256, 0, stream>>>(We, Webf);

    // 1. Encoder GEMM (bf16 WMMA, async double-buffered LDS): enc <- dense emb.
    enc_gemm<<<dim3(OUTF / 128, BATCH / 128), 256, 32768, stream>>>(xbf, Webf, be, enc);

    // 2. Transpose Wd for coalesced sparse gathers (independent of 1).
    transpose_wd<<<dim3(OUTF / 32, INF_ / 32), dim3(32, 8), 0, stream>>>(Wd, WdT);

    // 3. Exact per-row top-128 select + in-place mask + compaction.
    topk_mask<<<BATCH, 256, (size_t)OUTF * sizeof(unsigned), stream>>>(enc, wsIdx, wsVal);

    // 4. Sparse decode (192x fewer FLOPs than dense).
    dec_sparse<<<BATCH, 256, 0, stream>>>(wsIdx, wsVal, WdT, bd, dec);

    // 5. Third output is x itself.
    hipMemcpyAsync(xcp, x, (size_t)BATCH * INF_ * sizeof(float),
                   hipMemcpyDeviceToDevice, stream);
}
